// GraphAttention_20547123544359
// MI455X (gfx1250) — matrix operand order repaired and checked by reading, not compiled, not run on hardware
//
#include <hip/hip_runtime.h>
#include <hip/hip_bf16.h>

// Problem constants (from reference)
#define B_  32
#define N_  8192
#define D_  384        // NODE_DIM + 2*LATENT_DIM
#define L_  128        // LATENT_DIM
#define TN  64         // h-rows per block (LDS tile: 64*384*2B = 48KB)
#define CHUNKS (N_/TN) // 128
#define PS (2 + D_)    // floats per chunk partial: m, s, w[384]

typedef __attribute__((ext_vector_type(16))) __bf16 v16bf;
typedef __attribute__((ext_vector_type(8)))  float  v8f;

struct U4 { unsigned x, y, z, w; };

union ABReg {                 // one WMMA bf16 operand (8 VGPRs / 16 bf16 per lane)
  v16bf    v;
  unsigned u[8];
  U4       q[2];
};

__device__ __forceinline__ unsigned short f2bf(float f) {   // f32 -> bf16 RNE (prep only)
  unsigned x = __float_as_uint(f);
  x += 0x7FFFu + ((x >> 16) & 1u);
  return (unsigned short)(x >> 16);
}
// Pack hi16(a), hi16(b) -> one dword in a single v_perm_b32 (bf16 truncation)
__device__ __forceinline__ unsigned pk2t(float a, float b) {
  return __builtin_amdgcn_perm(__float_as_uint(b), __float_as_uint(a), 0x07060302u);
}
__device__ __forceinline__ float bf2f(unsigned short s) {
  return __uint_as_float(((unsigned)s) << 16);
}

// ---------------------------------------------------------------------------
// Kernel 0: per batch row, q = target@Wq + bq ; u = Wk@q (stored bf16-packed);
// cq = q.bk
// ---------------------------------------------------------------------------
__global__ __launch_bounds__(128)
void prep_kernel(const float* __restrict__ target, const float* __restrict__ Wq,
                 const float* __restrict__ bq, const float* __restrict__ Wk,
                 const float* __restrict__ bk, unsigned short* __restrict__ ubf,
                 float* __restrict__ cq) {
  const int b = blockIdx.x, t = threadIdx.x;
  __shared__ float ts[D_];
  __shared__ float qs[L_];
  __shared__ float rs[L_];
  for (int i = t; i < D_; i += 128) ts[i] = target[b * D_ + i];
  __syncthreads();
  float acc = bq[t];
  for (int d = 0; d < D_; ++d) acc += ts[d] * Wq[d * L_ + t];
  qs[t] = acc;
  rs[t] = acc * bk[t];
  __syncthreads();
  if (t == 0) { float s = 0.f; for (int j = 0; j < L_; ++j) s += rs[j]; cq[b] = s; }
  for (int i = t; i < D_; i += 128) {
    float a = 0.f;
    for (int j = 0; j < L_; ++j) a += Wk[i * L_ + j] * qs[j];
    ubf[b * D_ + i] = f2bf(a);          // B operand pre-converted to bf16 once
  }
}

// ---------------------------------------------------------------------------
// Kernel 1: one block = one (b, 64-row chunk of h).
//  Phase 1: stream h chunk (fp32 b128 loads), v_perm-pack to bf16, stage in
//           LDS, and accumulate scores = h_tile @ u via
//           v_wmma_f32_16x16x32_bf16 (u replicated across all 16 B-columns).
//  Phase 2: chunk-local softmax partials (m, s) + unnormalized Σ e[n]*h[n,:]
//           read back from the LDS bf16 tile (h touched in HBM exactly once).
// ---------------------------------------------------------------------------
__global__ __launch_bounds__(128)
void attn_chunk_kernel(const float* __restrict__ h,
                       const unsigned char* __restrict__ mask,  // jnp bool = 1 byte
                       const unsigned short* __restrict__ ubf,  // [B, D] bf16
                       const float* __restrict__ cq,
                       float* __restrict__ part) {
  const int chunk = blockIdx.x, b = blockIdx.y;
  const int t    = threadIdx.x;
  const int lane = t & 31;          // wave32
  const int wave = t >> 5;          // 4 waves per block
  const int hv   = lane >> 4;       // half-wave id (A/B K-group select)
  const int n0   = chunk * TN;
  const int wrow = wave * 16 + (lane & 15);   // tile row this lane loads A for

  __shared__ __align__(16) unsigned short hb[TN][D_];  // bf16 h tile, 48KB
  __shared__ float ev[TN];

  const float*          hrow = h   + ((size_t)b * N_ + n0 + wrow) * D_;
  const unsigned short* ub   = ubf + b * D_;

  v8f acc = {};
  #pragma unroll 4
  for (int kk = 0; kk < 12; ++kk) {           // K = 384 in steps of 32
    const int K0 = kk * 32;
    const int ka = K0 + hv * 8;               // A: elems 0..7 -> K=ka+j ; 8..15 -> K=ka+16+j
    if (kk + 1 < 12) __builtin_prefetch(hrow + K0 + 32, 0, 1);  // global_prefetch_b8
    float4 a0 = *reinterpret_cast<const float4*>(hrow + ka);
    float4 a1 = *reinterpret_cast<const float4*>(hrow + ka + 4);
    float4 a2 = *reinterpret_cast<const float4*>(hrow + ka + 16);
    float4 a3 = *reinterpret_cast<const float4*>(hrow + ka + 20);
    ABReg A;
    A.u[0] = pk2t(a0.x, a0.y); A.u[1] = pk2t(a0.z, a0.w);
    A.u[2] = pk2t(a1.x, a1.y); A.u[3] = pk2t(a1.z, a1.w);
    A.u[4] = pk2t(a2.x, a2.y); A.u[5] = pk2t(a2.z, a2.w);
    A.u[6] = pk2t(a3.x, a3.y); A.u[7] = pk2t(a3.z, a3.w);
    // stage the converted tile row-major in LDS for phase 2 (2x ds_store_b128)
    *reinterpret_cast<U4*>(&hb[wrow][ka])      = A.q[0];
    *reinterpret_cast<U4*>(&hb[wrow][ka + 16]) = A.q[1];
    // B operand: same K order as A (elems 0..7 -> K=ka+j ; 8..15 -> K=ka+16+j),
    // identical for every column (pre-converted bf16 u)
    ABReg Bm;
    const U4* up = reinterpret_cast<const U4*>(ub + K0 + hv * 8);
    Bm.q[0] = up[0];
    Bm.q[1] = up[2];
    acc = __builtin_amdgcn_wmma_f32_16x16x32_bf16(false, A.v, false, Bm.v,
                                                  (short)0, acc, false, false);
  }

  // C layout: lane (lane&15)==0 holds rows wave*16 + hv*8 + r in acc[r]
  const float scale = 0.05103103630798288f;   // 1/sqrt(384)
  const float cqb = cq[b];
  if ((lane & 15) == 0) {
    #pragma unroll
    for (int r = 0; r < 8; ++r) {
      const int row = wave * 16 + hv * 8 + r;
      float sc = (acc[r] + cqb) * scale;
      if (!mask[(size_t)b * N_ + n0 + row]) sc = -1e30f;
      ev[row] = sc;
    }
  }
  __syncthreads();
  float m = -1e30f;
  for (int i = 0; i < TN; ++i) m = fmaxf(m, ev[i]);   // broadcast LDS reads
  __syncthreads();
  if (t < TN) {
    const float sc = ev[t];
    ev[t] = (sc < -1e29f) ? 0.f : __expf(sc - m);     // all-masked chunk -> 0
  }
  __syncthreads();
  float ssum = 0.f;
  for (int i = 0; i < TN; ++i) ssum += ev[i];

  // phase 2: w[f] = sum_r e[r] * h_bf16[r][f]; thread owns f = 2t, 2t+1, 256+t
  float w0 = 0.f, w1 = 0.f, w2 = 0.f;
  for (int r = 0; r < TN; ++r) {
    const float e = ev[r];
    const unsigned d01 = *reinterpret_cast<const unsigned*>(&hb[r][2 * t]); // ds_load_b32
    w0 += e * bf2f((unsigned short)(d01 & 0xFFFFu));
    w1 += e * bf2f((unsigned short)(d01 >> 16));
    w2 += e * bf2f(hb[r][256 + t]);
  }
  float* p = part + ((size_t)b * CHUNKS + chunk) * PS;
  if (t == 0) { p[0] = m; p[1] = ssum; }
  p[2 + 2 * t]     = w0;
  p[2 + 2 * t + 1] = w1;
  p[2 + 256 + t]   = w2;
}

// ---------------------------------------------------------------------------
// Kernel 2: per-b LSE combine of 128 chunk partials, then
// nws = (w/S)@Wv + bv ; z = relu([nws,target]@W1 + b1)@W2 + b2
// ---------------------------------------------------------------------------
__global__ __launch_bounds__(128)
void finalize_kernel(const float* __restrict__ part,
                     const float* __restrict__ target,
                     const float* __restrict__ Wv, const float* __restrict__ bv,
                     const float* __restrict__ W1, const float* __restrict__ b1,
                     const float* __restrict__ W2, const float* __restrict__ b2,
                     float* __restrict__ out) {
  const int b = blockIdx.x, t = threadIdx.x;
  const float* pb = part + (size_t)b * CHUNKS * PS;
  __shared__ float cs[CHUNKS];
  __shared__ float wsum[D_];
  __shared__ float zc[L_ + D_];
  __shared__ float z1s[L_];

  float M = -1e30f;
  for (int c = 0; c < CHUNKS; ++c) M = fmaxf(M, pb[(size_t)c * PS]);
  {
    const float mc = pb[(size_t)t * PS];      // CHUNKS == blockDim == 128
    cs[t] = (mc < -1e29f) ? 0.f : __expf(mc - M);
  }
  __syncthreads();
  float S = 0.f;
  for (int c = 0; c < CHUNKS; ++c) S += cs[c] * pb[(size_t)c * PS + 1];
  for (int f = t; f < D_; f += 128) {
    float a = 0.f;
    for (int c = 0; c < CHUNKS; ++c) a += cs[c] * pb[(size_t)c * PS + 2 + f];
    wsum[f] = a;
  }
  __syncthreads();
  const float invS = 1.0f / S;
  float a = 0.f;
  for (int f = 0; f < D_; ++f) a += wsum[f] * Wv[f * L_ + t];
  zc[t] = a * invS + bv[t];                              // nws (first 128 of concat)
  for (int i = t; i < D_; i += 128) zc[L_ + i] = target[b * D_ + i];
  __syncthreads();
  float a1 = b1[t];
  for (int i = 0; i < L_ + D_; ++i) a1 += zc[i] * W1[i * L_ + t];
  z1s[t] = fmaxf(a1, 0.f);
  __syncthreads();
  float a2 = b2[t];
  for (int j = 0; j < L_; ++j) a2 += z1s[j] * W2[j * L_ + t];
  out[(size_t)b * L_ + t] = a2;
}

// ---------------------------------------------------------------------------
extern "C" void kernel_launch(void* const* d_in, const int* in_sizes, int n_in,
                              void* d_out, int out_size, void* d_ws, size_t ws_size,
                              hipStream_t stream) {
  const float* target = (const float*)d_in[0];
  const float* h      = (const float*)d_in[1];
  const unsigned char* mask = (const unsigned char*)d_in[2];  // bool -> 1 byte
  const float* Wq = (const float*)d_in[3];
  const float* bq = (const float*)d_in[4];
  const float* Wk = (const float*)d_in[5];
  const float* bk = (const float*)d_in[6];
  const float* Wv = (const float*)d_in[7];
  const float* bv = (const float*)d_in[8];
  const float* W1 = (const float*)d_in[9];
  const float* b1 = (const float*)d_in[10];
  const float* W2 = (const float*)d_in[11];
  const float* b2 = (const float*)d_in[12];
  float* out = (float*)d_out;

  // workspace layout (~6.4 MB)
  unsigned short* ubf = (unsigned short*)d_ws;           // [B, D] bf16 (Wk @ q)
  float* cq   = (float*)((char*)d_ws + B_ * D_ * sizeof(unsigned short)); // [B]
  float* part = cq + B_;                                 // [B, CHUNKS, PS]

  prep_kernel<<<B_, 128, 0, stream>>>(target, Wq, bq, Wk, bk, ubf, cq);
  attn_chunk_kernel<<<dim3(CHUNKS, B_), 128, 0, stream>>>(h, mask, ubf, cq, part);
  finalize_kernel<<<B_, 128, 0, stream>>>(part, target, Wv, bv, W1, b1, W2, b2, out);
}
